// MultiheadAttention_54391465836906
// MI455X (gfx1250) — matrix operand order repaired and checked by reading, not compiled, not run on hardware
//
#include <hip/hip_runtime.h>
#include <hip/hip_bf16.h>
#include <stdint.h>

// ---------------------------------------------------------------------------
// Problem constants (B=2, S=2048, E=1024, H=16, DH=64)
// ---------------------------------------------------------------------------
#define kB 2
#define kS 2048
#define kE 1024
#define kH 16
#define kD 64
#define kM (kB * kS) /* 4096 rows of the (B*S) x E activation matrix */

// ---------------------------------------------------------------------------
// WMMA fragment types (CDNA5 wave32; v_wmma_f32_16x16x32_bf16)
// ---------------------------------------------------------------------------
typedef __attribute__((ext_vector_type(16))) __bf16 v16bf;
typedef __attribute__((ext_vector_type(8))) float v8f;

union Frag {
  unsigned int u[8];  // u32 view: VGPR i holds bf16 pair (lo, hi)
  v16bf b;
};

static __device__ __forceinline__ unsigned short f2bf(float f) {
  unsigned int u = __float_as_uint(f);
  u += 0x7FFFu + ((u >> 16) & 1u);  // round-to-nearest-even
  return (unsigned short)(u >> 16);
}

static __device__ __forceinline__ v8f wmma_bf16(const Frag& a, const Frag& b,
                                                v8f c) {
  return __builtin_amdgcn_wmma_f32_16x16x32_bf16(false, a.b, false, b.b,
                                                 (short)0, c, false, false);
}

// ---------------------------------------------------------------------------
// CDNA5 async memory->LDS staging (no VGPR transit; tracked by ASYNCcnt).
// Builtin signature (from clang diagnostic): param0 = v4i addrspace(1)*,
// param1 = v4i addrspace(3)*, then two immediate ints (offset, cpol).
// Casts go through uintptr_t to avoid addrspace-conversion legality issues;
// the LDS pointer truncates to 32 bits.
// ---------------------------------------------------------------------------
#if __has_builtin(__builtin_amdgcn_global_load_async_to_lds_b128)
#define USE_ASYNC_TILE_LOAD 1
#else
#define USE_ASYNC_TILE_LOAD 0
#endif

typedef __attribute__((__vector_size__(4 * sizeof(int)))) int v4i;
#define ASG4(p) ((__attribute__((address_space(1))) v4i*)(uintptr_t)(p))
#define ASL4(p) \
  ((__attribute__((address_space(3))) v4i*)(unsigned int)(uintptr_t)(p))

static __device__ __forceinline__ void wait_async0() {
#if __has_builtin(__builtin_amdgcn_s_wait_asynccnt)
  __builtin_amdgcn_s_wait_asynccnt(0);
#else
  asm volatile("s_wait_asynccnt 0x0" ::: "memory");
#endif
}

// ---------------------------------------------------------------------------
// 1) f32 -> bf16 conversion
// ---------------------------------------------------------------------------
__global__ void cvt_bf16_kernel(const float* __restrict__ src,
                                unsigned short* __restrict__ dst, int n) {
  int i = blockIdx.x * blockDim.x + threadIdx.x;
  if (i < n) dst[i] = f2bf(src[i]);
}

// ---------------------------------------------------------------------------
// 2) Generic bf16 WMMA GEMM:  C[m,n] = sum_k A[m,k] * W[n,k]
//    A: [kM][kE] bf16 row-major (viewed as u32 pairs), W: [kE][kE] bf16.
//    Block: 256 thr (8 waves), 128x128 tile, K stepped by 32 through LDS.
//    Epilogue modes:
//      0: RoPE, store Q  bf16 [b,h,s,d]
//      1: RoPE, store K  bf16 [b,h,s,d]
//      2: no RoPE, store V^T bf16 [b,h,d,s]
//      3: add bias, store f32 [m][kE]  (final projection)
// ---------------------------------------------------------------------------
__global__ __launch_bounds__(256) void gemm_rope_kernel(
    const unsigned int* __restrict__ Ag, const unsigned int* __restrict__ Bg,
    int mode, unsigned short* __restrict__ outb, float* __restrict__ outf,
    const float* __restrict__ bias) {
  __shared__ unsigned int Asm[128 * 16];  // 128 rows x 32 bf16 (16 u32 pairs)
  __shared__ unsigned int Bsm[128 * 16];

  const int t = threadIdx.x;
  const int w = t >> 5;
  const int lane = t & 31;
  const int lo = lane & 15;
  const int hi = lane >> 4;

  const int m0 = blockIdx.x * 128;
  const int n0 = blockIdx.y * 128;

  v8f acc[8];
#pragma unroll
  for (int nt = 0; nt < 8; ++nt)
#pragma unroll
    for (int r = 0; r < 8; ++r) acc[nt][r] = 0.0f;

  for (int kc = 0; kc < kE / 32; ++kc) {
    const int kp0 = kc * 16;  // u32 (bf16-pair) offset within a row
    __syncthreads();
#pragma unroll
    for (int c = 0; c < 2; ++c) {
      const int chunk = t + c * 256;  // 512 x 16B chunks per matrix
      const int row = chunk >> 2;
      const int q4 = (chunk & 3) * 4;
      const unsigned int* ga = &Ag[(size_t)(m0 + row) * (kE / 2) + kp0 + q4];
      const unsigned int* gb = &Bg[(size_t)(n0 + row) * (kE / 2) + kp0 + q4];
      unsigned int* la = &Asm[row * 16 + q4];
      unsigned int* lb = &Bsm[row * 16 + q4];
#if USE_ASYNC_TILE_LOAD
      __builtin_amdgcn_global_load_async_to_lds_b128(ASG4(ga), ASL4(la), 0, 0);
      __builtin_amdgcn_global_load_async_to_lds_b128(ASG4(gb), ASL4(lb), 0, 0);
#else
      *(uint4*)la = *(const uint4*)ga;
      *(uint4*)lb = *(const uint4*)gb;
#endif
      if (kc + 1 < kE / 32) {  // L2 prefetch of next K step (global_prefetch)
        __builtin_prefetch(ga + 16, 0, 3);
        __builtin_prefetch(gb + 16, 0, 3);
      }
    }
#if USE_ASYNC_TILE_LOAD
    wait_async0();
#endif
    __syncthreads();

    // A fragment for this wave's 16-row strip (contiguous k order, same as B:
    // lane lo = row; pair index p = hi*8 + i).
    Frag fa;
#pragma unroll
    for (int i = 0; i < 8; ++i) {
      const int p = hi * 8 + i;
      fa.u[i] = Asm[(w * 16 + lo) * 16 + p];
    }
#pragma unroll
    for (int nt = 0; nt < 8; ++nt) {
      // B fragment (contiguous k order, same as A: lane lo = col n; VGPR i holds
      // K = hi*16 + 2i, 2i+1; Bsm rows are W rows = Bt rows).
      Frag fb;
#pragma unroll
      for (int i = 0; i < 8; ++i)
        fb.u[i] = Bsm[(nt * 16 + lo) * 16 + hi * 8 + i];
      acc[nt] = wmma_bf16(fa, fb, acc[nt]);
    }
  }

  // ---- epilogue ----
  const int mB = m0 + w * 16;
#pragma unroll
  for (int nt = 0; nt < 8; ++nt) {
    const int n = n0 + nt * 16 + lo;  // this lane's output column
    if (mode == 3) {
      const float bv = bias[n];
#pragma unroll
      for (int r = 0; r < 8; ++r) {
        const int m = mB + r + 8 * hi;
        outf[(size_t)m * kE + n] = acc[nt][r] + bv;
      }
    } else if (mode == 2) {
      const int h = n >> 6, d = n & 63;
#pragma unroll
      for (int r = 0; r < 8; ++r) {
        const int m = mB + r + 8 * hi;
        const int bh = (m >> 11) * kH + h;
        const int s = m & (kS - 1);
        outb[((size_t)bh * kD + d) * kS + s] = f2bf(acc[nt][r]);  // V^T
      }
    } else {
      // RoPE: pair partner column d^1 lives in lane lo^1 of the C layout.
      const int h = n >> 6, d = n & 63;
      const float theta =
          __expf(-(float)(d & ~1) * (9.210340371976184f / (float)kD));
      const float sgn = (d & 1) ? 1.0f : -1.0f;
#pragma unroll
      for (int r = 0; r < 8; ++r) {
        const int m = mB + r + 8 * hi;
        const int bh = (m >> 11) * kH + h;
        const int s = m & (kS - 1);
        const float ang = (float)s * theta;
        const float self = acc[nt][r];
        const float partner = __shfl_xor(self, 1, 32);
        const float ro = self * __cosf(ang) + sgn * partner * __sinf(ang);
        outb[((size_t)bh * kS + s) * kD + d] = f2bf(ro);
      }
    }
  }
}

// ---------------------------------------------------------------------------
// 3) Causal flash attention, one wave per 16-row q tile.
//    Q: [bh][s][d] bf16, K: [bh][s][d] bf16, V^T: [bh][d][s] bf16.
//    Output: [b*S][E] bf16 (feeds final projection GEMM).
// ---------------------------------------------------------------------------
static __device__ __forceinline__ v8f score_tile(const Frag* aQ,
                                                 const unsigned int* krow,
                                                 int hi) {
  Frag b0, b1;
#pragma unroll
  for (int i = 0; i < 8; ++i) {
    b0.u[i] = krow[hi * 8 + i];       // d = 0..31
    b1.u[i] = krow[16 + hi * 8 + i];  // d = 32..63
  }
  v8f s;
#pragma unroll
  for (int r = 0; r < 8; ++r) s[r] = 0.0f;
  s = wmma_bf16(aQ[0], b0, s);
  s = wmma_bf16(aQ[1], b1, s);
  return s;
}

__global__ __launch_bounds__(256) void attn_wmma_kernel(
    const unsigned int* __restrict__ Qg, const unsigned int* __restrict__ Kg,
    const unsigned int* __restrict__ Vtg, unsigned short* __restrict__ Og) {
  __shared__ unsigned short Pst[8][16 * 32];  // per-wave P staging (C->A)

  const int t = threadIdx.x;
  const int w = t >> 5;
  const int lane = t & 31;
  const int lo = lane & 15;
  const int hi = lane >> 4;

  const int bh = blockIdx.y;
  const int b = bh >> 4;
  const int h = bh & 15;
  const int qt = blockIdx.x * 8 + w;
  const int q0 = qt * 16;

  // Q fragments (A layout), loaded once and reused across all kv chunks.
  Frag aQ[2];
  {
    const unsigned int* qrow = Qg + ((size_t)bh * kS + (q0 + lo)) * 32;
#pragma unroll
    for (int half = 0; half < 2; ++half)
#pragma unroll
      for (int i = 0; i < 8; ++i) {
        const int p = hi * 8 + i;
        aQ[half].u[i] = qrow[half * 16 + p];
      }
  }

  float mi[8], li[8];
#pragma unroll
  for (int r = 0; r < 8; ++r) {
    mi[r] = -1e30f;
    li[r] = 0.0f;
  }
  v8f Oacc[4];
#pragma unroll
  for (int tt = 0; tt < 4; ++tt)
#pragma unroll
    for (int r = 0; r < 8; ++r) Oacc[tt][r] = 0.0f;

  const float sc = 0.125f;  // 1/sqrt(DH)
  const int nchunks = (q0 + 47) >> 5;
  unsigned int* Pw32 = (unsigned int*)&Pst[w][0];

  for (int kc = 0; kc < nchunks; ++kc) {
    const int kv0 = kc << 5;
    float z0[8], z1[8];

    {  // sub-tile 0: kv = kv0 .. kv0+15
      const unsigned int* krow = Kg + ((size_t)bh * kS + (kv0 + lo)) * 32;
      v8f s = score_tile(aQ, krow, hi);
#pragma unroll
      for (int r = 0; r < 8; ++r) {
        const int sq = q0 + r + 8 * hi;
        z0[r] = ((kv0 + lo) > sq) ? -1e20f : s[r];  // mask BEFORE scaling
      }
    }
    const int kv16 = kv0 + 16;
    if (kv16 <= q0 + 15) {  // sub-tile 1 (wave-uniform; EXEC stays full)
      const unsigned int* krow = Kg + ((size_t)bh * kS + (kv16 + lo)) * 32;
      v8f s = score_tile(aQ, krow, hi);
#pragma unroll
      for (int r = 0; r < 8; ++r) {
        const int sq = q0 + r + 8 * hi;
        z1[r] = ((kv16 + lo) > sq) ? -1e20f : s[r];
      }
    } else {
#pragma unroll
      for (int r = 0; r < 8; ++r) z1[r] = -1e30f;
    }

    // Online softmax: row stats live replicated across each 16-lane half.
    float a[8], p0[8], p1[8];
#pragma unroll
    for (int r = 0; r < 8; ++r) {
      float rm = fmaxf(z0[r], z1[r]);
      rm = fmaxf(rm, __shfl_xor(rm, 1, 32));
      rm = fmaxf(rm, __shfl_xor(rm, 2, 32));
      rm = fmaxf(rm, __shfl_xor(rm, 4, 32));
      rm = fmaxf(rm, __shfl_xor(rm, 8, 32));
      const float mnew = fmaxf(mi[r], rm);
      a[r] = __expf((mi[r] - mnew) * sc);
      p0[r] = __expf((z0[r] - mnew) * sc);
      p1[r] = __expf((z1[r] - mnew) * sc);
      float pr = p0[r] + p1[r];
      pr += __shfl_xor(pr, 1, 32);
      pr += __shfl_xor(pr, 2, 32);
      pr += __shfl_xor(pr, 4, 32);
      pr += __shfl_xor(pr, 8, 32);
      li[r] = li[r] * a[r] + pr;
      mi[r] = mnew;
    }
#pragma unroll
    for (int tt = 0; tt < 4; ++tt)
#pragma unroll
      for (int r = 0; r < 8; ++r) Oacc[tt][r] *= a[r];

    // Stage P (C layout) to LDS as bf16, re-gather in A layout.
#pragma unroll
    for (int r = 0; r < 8; ++r) {
      const int row = r + 8 * hi;
      Pst[w][row * 32 + lo] = f2bf(p0[r]);
      Pst[w][row * 32 + 16 + lo] = f2bf(p1[r]);
    }
    Frag aP;
#pragma unroll
    for (int i = 0; i < 8; ++i) {
      const int p = hi * 8 + i;
      aP.u[i] = Pw32[lo * 16 + p];
    }

    // O += P @ V : V^T rows are contiguous -> direct B-layout global gathers.
#pragma unroll
    for (int tt = 0; tt < 4; ++tt) {
      const int d = tt * 16 + lo;
      const unsigned int* vrow =
          Vtg + ((size_t)bh * kD + d) * (kS / 2) + (kv0 >> 1);
      Frag bv;
#pragma unroll
      for (int i = 0; i < 8; ++i) bv.u[i] = vrow[hi * 8 + i];
      Oacc[tt] = wmma_bf16(aP, bv, Oacc[tt]);
    }
  }

  // Normalize and store to [b*S][E] bf16 for the output projection GEMM.
#pragma unroll
  for (int r = 0; r < 8; ++r) li[r] = 1.0f / li[r];
#pragma unroll
  for (int tt = 0; tt < 4; ++tt) {
    const int d = tt * 16 + lo;
#pragma unroll
    for (int r = 0; r < 8; ++r) {
      const int s = q0 + r + 8 * hi;
      Og[((size_t)b * kS + s) * kE + h * kD + d] = f2bf(Oacc[tt][r] * li[r]);
    }
  }
}

// ---------------------------------------------------------------------------
// Host launcher
// ---------------------------------------------------------------------------
extern "C" void kernel_launch(void* const* d_in, const int* in_sizes, int n_in,
                              void* d_out, int out_size, void* d_ws,
                              size_t ws_size, hipStream_t stream) {
  (void)in_sizes;
  (void)n_in;
  (void)out_size;
  (void)ws_size;

  const float* x = (const float*)d_in[0];
  // d_in[1] = causal mask (int32): known analytically, not read on device.
  const float* Wq = (const float*)d_in[2];
  const float* Wk = (const float*)d_in[3];
  const float* Wv = (const float*)d_in[4];
  const float* Wo = (const float*)d_in[5];
  const float* bo = (const float*)d_in[6];
  float* out = (float*)d_out;

  // Workspace layout (bf16 buffers), total = 48 MB.
  char* ws = (char*)d_ws;
  size_t off = 0;
  unsigned short* xbf = (unsigned short*)(ws + off);
  off += (size_t)kM * kE * 2;  //  8 MB
  unsigned short* wqb = (unsigned short*)(ws + off);
  off += (size_t)kE * kE * 2;  //  2 MB
  unsigned short* wkb = (unsigned short*)(ws + off);
  off += (size_t)kE * kE * 2;
  unsigned short* wvb = (unsigned short*)(ws + off);
  off += (size_t)kE * kE * 2;
  unsigned short* wob = (unsigned short*)(ws + off);
  off += (size_t)kE * kE * 2;
  unsigned short* qbuf = (unsigned short*)(ws + off);
  off += (size_t)kB * kH * kS * kD * 2;  //  8 MB
  unsigned short* kbuf = (unsigned short*)(ws + off);
  off += (size_t)kB * kH * kS * kD * 2;
  unsigned short* vtbuf = (unsigned short*)(ws + off);
  off += (size_t)kB * kH * kS * kD * 2;
  unsigned short* abuf = (unsigned short*)(ws + off);
  off += (size_t)kM * kE * 2;

  // 1) convert fp32 -> bf16
  const int nX = kM * kE;
  const int nW = kE * kE;
  cvt_bf16_kernel<<<(nX + 255) / 256, 256, 0, stream>>>(x, xbf, nX);
  cvt_bf16_kernel<<<(nW + 255) / 256, 256, 0, stream>>>(Wq, wqb, nW);
  cvt_bf16_kernel<<<(nW + 255) / 256, 256, 0, stream>>>(Wk, wkb, nW);
  cvt_bf16_kernel<<<(nW + 255) / 256, 256, 0, stream>>>(Wv, wvb, nW);
  cvt_bf16_kernel<<<(nW + 255) / 256, 256, 0, stream>>>(Wo, wob, nW);

  // 2) Q/K/V projections (+RoPE on Q,K; V stored transposed)
  dim3 gg(kM / 128, kE / 128);  // 32 x 8
  dim3 blk(256);
  gemm_rope_kernel<<<gg, blk, 0, stream>>>((const unsigned int*)xbf,
                                           (const unsigned int*)wqb, 0, qbuf,
                                           nullptr, nullptr);
  gemm_rope_kernel<<<gg, blk, 0, stream>>>((const unsigned int*)xbf,
                                           (const unsigned int*)wkb, 1, kbuf,
                                           nullptr, nullptr);
  gemm_rope_kernel<<<gg, blk, 0, stream>>>((const unsigned int*)xbf,
                                           (const unsigned int*)wvb, 2, vtbuf,
                                           nullptr, nullptr);

  // 3) causal flash attention (one wave per 16-row q tile)
  dim3 ga(kS / 128, kB * kH);  // 16 x 32 blocks, 8 waves each
  attn_wmma_kernel<<<ga, blk, 0, stream>>>(
      (const unsigned int*)qbuf, (const unsigned int*)kbuf,
      (const unsigned int*)vtbuf, abuf);

  // 4) output projection + bias -> fp32 d_out
  gemm_rope_kernel<<<gg, blk, 0, stream>>>((const unsigned int*)abuf,
                                           (const unsigned int*)wob, 3, nullptr,
                                           out, bo);
}
